// BertCRFTokenClassModel_37735582663453
// MI455X (gfx1250) — compile-verified, matrix operand order checked by reading, not run on hardware
//
#include <hip/hip_runtime.h>

// ---------------- model constants ----------------
#define B_    32
#define T_    256
#define H_    768
#define L_    12
#define NH_   12
#define DH_   64
#define FF_   3072
#define NL_   31
#define NLP_  32          // padded feats row width
#define START_ 21
#define NEG_  -10000.0f

typedef __bf16 bf16;
typedef __attribute__((ext_vector_type(8)))  bf16  v8bf;
typedef __attribute__((ext_vector_type(16))) bf16  v16bf;
typedef __attribute__((ext_vector_type(8)))  float v8f;

// -------------------------------------------------------------------------
// WMMA fragment load, CDNA5 16-bit A/B layout (16x32 / 32x16):
//   lane l: row/col = l&15, half = l>>4
//   elements 0..7  -> K = half*8 + 0..7        (contiguous 16B)
//   elements 8..15 -> K = 16 + half*8 + 0..7   (contiguous 16B)
// p points at row start + k0 (16B aligned in all our uses).
// -------------------------------------------------------------------------
__device__ __forceinline__ v16bf load_frag(const bf16* p, int half) {
    v8bf a = *(const v8bf*)(p + half * 8);
    v8bf b = *(const v8bf*)(p + 16 + half * 8);
    return __builtin_shufflevector(a, b, 0,1,2,3,4,5,6,7,8,9,10,11,12,13,14,15);
}

// -------------------------------------------------------------------------
// Generic WMMA GEMM:  C[M,N] = act( alpha * A[M,K] * Bt[N,K]^T + bias[N] )
// A, Bt row-major along K (lda/ldb elem strides). Batched over blockIdx.z with
// nonlinear offset decomposition: off = (z/zdiv)*Outer + (z%zdiv)*Inner.
// Block = 256 threads = 8 waves arranged 2(M) x 4(N).
// Wave tile = 64x64 (4x4 WMMA tiles) -> 32 FLOP/byte of fragment traffic.
// Workgroup tile = 128(M) x 256(N).
// -------------------------------------------------------------------------
__global__ __launch_bounds__(256) void gemm_bf16(
    const bf16* __restrict__ A, const bf16* __restrict__ Bt,
    const float* __restrict__ bias,
    float* __restrict__ C32, bf16* __restrict__ C16,
    int M, int N, int K, int lda, int ldb, int ldc,
    int zdiv,
    long long aO, long long aI, long long bO, long long bI,
    long long cO, long long cI,
    float alpha, int act)
{
    const int tid  = threadIdx.x;
    const int lane = tid & 31;
    const int wave = tid >> 5;
    const int ln   = lane & 15;
    const int half = lane >> 4;

    const int z  = blockIdx.z;
    const int zq = z / zdiv;
    const int zr = z - zq * zdiv;
    A  += (size_t)zq * aO + (size_t)zr * aI;
    Bt += (size_t)zq * bO + (size_t)zr * bI;
    const size_t coff = (size_t)zq * cO + (size_t)zr * cI;

    const int m0 = blockIdx.y * 128 + (wave >> 2) * 64;   // 2 waves in M
    const int n0 = blockIdx.x * 256 + (wave & 3) * 64;    // 4 waves in N

    const bf16* arow[4];
    const bf16* brow[4];
#pragma unroll
    for (int i = 0; i < 4; ++i) {
        int m = m0 + i * 16 + ln;
        arow[i] = A + (size_t)(m < M ? m : 0) * lda;
        int n = n0 + i * 16 + ln;
        brow[i] = Bt + (size_t)(n < N ? n : 0) * ldb;
    }

    v8f acc[4][4] = {};

    for (int k0 = 0; k0 < K; k0 += 32) {
        v16bf bfr[4];
#pragma unroll
        for (int j = 0; j < 4; ++j) bfr[j] = load_frag(brow[j] + k0, half);
#pragma unroll
        for (int i = 0; i < 4; ++i) {
            v16bf af = load_frag(arow[i] + k0, half);
            acc[i][0] = __builtin_amdgcn_wmma_f32_16x16x32_bf16(false, af, false, bfr[0], (short)0, acc[i][0], false, false);
            acc[i][1] = __builtin_amdgcn_wmma_f32_16x16x32_bf16(false, af, false, bfr[1], (short)0, acc[i][1], false, false);
            acc[i][2] = __builtin_amdgcn_wmma_f32_16x16x32_bf16(false, af, false, bfr[2], (short)0, acc[i][2], false, false);
            acc[i][3] = __builtin_amdgcn_wmma_f32_16x16x32_bf16(false, af, false, bfr[3], (short)0, acc[i][3], false, false);
        }
    }

    // C/D layout: VGPR r, lanes 0-15 -> M = r ; lanes 16-31 -> M = 8 + r ; N = lane&15
#pragma unroll
    for (int i = 0; i < 4; ++i) {
#pragma unroll
        for (int j = 0; j < 4; ++j) {
            int n = n0 + j * 16 + ln;
            if (n >= N) continue;
            float bv = bias ? bias[n] : 0.0f;
#pragma unroll
            for (int r = 0; r < 8; ++r) {
                int m = m0 + i * 16 + half * 8 + r;
                if (m >= M) continue;
                float v = acc[i][j][r] * alpha + bv;
                if (act == 1) v = 0.5f * v * (1.0f + erff(v * 0.70710678118654752f));
                size_t o = coff + (size_t)m * ldc + n;
                if (C32) C32[o] = v;
                if (C16) C16[o] = (bf16)v;
            }
        }
    }
}

// ---------------- fp32 [K,N] -> bf16 transposed [N,K], batched over z ------
__global__ __launch_bounds__(256) void cvt_transpose(
    const float* __restrict__ src, bf16* __restrict__ dst,
    int K, int N, long long sZ, long long dZ)
{
    size_t z = blockIdx.z;
    src += z * (size_t)sZ;
    dst += z * (size_t)dZ;
    size_t idx = (size_t)blockIdx.x * 256 + threadIdx.x;
    if (idx >= (size_t)K * N) return;
    int k = (int)(idx % K);
    int n = (int)(idx / K);
    dst[(size_t)n * K + k] = (bf16)src[(size_t)k * N + n];
}

// ---------------- embedding + LayerNorm (block per token) ------------------
__global__ __launch_bounds__(256) void embed_ln(
    const int* __restrict__ ids,
    const float* __restrict__ tok, const float* __restrict__ pos,
    const float* __restrict__ typ,
    const float* __restrict__ g, const float* __restrict__ be,
    float* __restrict__ x, bf16* __restrict__ xb)
{
    const int token = blockIdx.x;
    const int t = token % T_;
    const int id = ids[token];
    const int tid = threadIdx.x;
    __shared__ float red[256];

    float v[3];
    for (int i = 0; i < 3; ++i) {
        int h = tid + i * 256;
        v[i] = tok[(size_t)id * H_ + h] + pos[(size_t)t * H_ + h] + typ[h];
    }
    float s = v[0] + v[1] + v[2];
    red[tid] = s; __syncthreads();
    for (int o = 128; o > 0; o >>= 1) { if (tid < o) red[tid] += red[tid + o]; __syncthreads(); }
    float mu = red[0] / (float)H_; __syncthreads();
    float sq = 0.f;
    for (int i = 0; i < 3; ++i) { float d = v[i] - mu; sq += d * d; }
    red[tid] = sq; __syncthreads();
    for (int o = 128; o > 0; o >>= 1) { if (tid < o) red[tid] += red[tid + o]; __syncthreads(); }
    float rs = rsqrtf(red[0] / (float)H_ + 1e-12f);
    for (int i = 0; i < 3; ++i) {
        int h = tid + i * 256;
        float y = (v[i] - mu) * rs * g[h] + be[h];
        x [(size_t)token * H_ + h] = y;
        xb[(size_t)token * H_ + h] = (bf16)y;
    }
}

// ---------------- x = LN(x + tmp) (tmp already has bias) -------------------
__global__ __launch_bounds__(256) void residual_ln(
    float* __restrict__ x, const float* __restrict__ tmp,
    const float* __restrict__ g, const float* __restrict__ be,
    bf16* __restrict__ xb)
{
    const int token = blockIdx.x;
    const int tid = threadIdx.x;
    __shared__ float red[256];

    float v[3];
    for (int i = 0; i < 3; ++i) {
        size_t o = (size_t)token * H_ + tid + i * 256;
        v[i] = x[o] + tmp[o];
    }
    float s = v[0] + v[1] + v[2];
    red[tid] = s; __syncthreads();
    for (int o = 128; o > 0; o >>= 1) { if (tid < o) red[tid] += red[tid + o]; __syncthreads(); }
    float mu = red[0] / (float)H_; __syncthreads();
    float sq = 0.f;
    for (int i = 0; i < 3; ++i) { float d = v[i] - mu; sq += d * d; }
    red[tid] = sq; __syncthreads();
    for (int o = 128; o > 0; o >>= 1) { if (tid < o) red[tid] += red[tid + o]; __syncthreads(); }
    float rs = rsqrtf(red[0] / (float)H_ + 1e-12f);
    for (int i = 0; i < 3; ++i) {
        int h = tid + i * 256;
        size_t o = (size_t)token * H_ + h;
        float y = (v[i] - mu) * rs * g[h] + be[h];
        x[o] = y;
        xb[o] = (bf16)y;
    }
}

// ---------------- masked softmax over one score row ------------------------
__global__ __launch_bounds__(256) void softmax_mask(
    const float* __restrict__ sc, const int* __restrict__ mask,
    bf16* __restrict__ p)
{
    const size_t row = blockIdx.x;             // b*NH*T + h*T + t
    const int b = (int)(row / ((size_t)NH_ * T_));
    const int si = threadIdx.x;
    __shared__ float red[256];

    float v = sc[row * T_ + si] + (1.0f - (float)mask[(size_t)b * T_ + si]) * NEG_;
    red[si] = v; __syncthreads();
    for (int o = 128; o > 0; o >>= 1) { if (si < o) red[si] = fmaxf(red[si], red[si + o]); __syncthreads(); }
    float mx = red[0]; __syncthreads();
    float e = __expf(v - mx);
    red[si] = e; __syncthreads();
    for (int o = 128; o > 0; o >>= 1) { if (si < o) red[si] += red[si + o]; __syncthreads(); }
    p[row * T_ + si] = (bf16)(e / red[0]);
}

// ---------------- v [B,T,H] bf16 -> vt [B,NH,DH,T] bf16 --------------------
__global__ __launch_bounds__(256) void transpose_v(
    const bf16* __restrict__ v, bf16* __restrict__ vt)
{
    size_t idx = (size_t)blockIdx.x * 256 + threadIdx.x;
    if (idx >= (size_t)B_ * NH_ * DH_ * T_) return;
    int s = (int)(idx % T_);
    size_t r = idx / T_;
    int d = (int)(r % DH_);
    size_t zz = r / DH_;
    int h = (int)(zz % NH_);
    int b = (int)(zz / NH_);
    vt[idx] = v[((size_t)b * T_ + s) * H_ + h * DH_ + d];
}

// ---------------- Viterbi decode: 1 block (1 wave) per batch ---------------
__global__ __launch_bounds__(32) void viterbi_kernel(
    const float* __restrict__ feats,      // [B*T, NLP_]
    const float* __restrict__ tr,         // [NL_,NL_]
    float* __restrict__ out,              // score[B_] then path[B_,T_]
    int* __restrict__ psi)                // [(T_-1), B_, NL_]
{
    const int b = blockIdx.x;
    const int j = threadIdx.x;
    __shared__ float trs[NL_ * NL_];
    __shared__ float sh[32];
    for (int i = j; i < NL_ * NL_; i += 32) trs[i] = tr[i];
    float ld = (j == START_) ? 0.0f : NEG_;
    __syncthreads();

    for (int t = 1; t < T_; ++t) {
        sh[j] = ld;
        __syncthreads();
        float nv = ld;
        if (j < NL_) {
            float best = -3.4e38f; int arg = 0;
            for (int i = 0; i < NL_; ++i) {
                float vv = trs[j * NL_ + i] + sh[i];
                if (vv > best) { best = vv; arg = i; }
            }
            nv = best + feats[((size_t)b * T_ + t) * NLP_ + j];
            psi[((size_t)(t - 1) * B_ + b) * NL_ + j] = arg;
        }
        __syncthreads();
        ld = nv;
    }

    sh[j] = (j < NL_) ? ld : -3.4e38f;
    __syncthreads();
    if (j == 0) {
        float best = sh[0]; int last = 0;
        for (int i = 1; i < NL_; ++i) if (sh[i] > best) { best = sh[i]; last = i; }
        out[b] = best;
        float* path = out + B_ + (size_t)b * T_;
        path[T_ - 1] = (float)last;
        int p = last;
        for (int t = T_ - 2; t >= 0; --t) {
            p = psi[((size_t)t * B_ + b) * NL_ + p];
            path[t] = (float)p;
        }
    }
}

// ===========================================================================
extern "C" void kernel_launch(void* const* d_in, const int* in_sizes, int n_in,
                              void* d_out, int out_size, void* d_ws, size_t ws_size,
                              hipStream_t stream) {
    (void)in_sizes; (void)n_in; (void)out_size; (void)ws_size;

    const int*   ids   = (const int*)  d_in[0];
    const int*   msk   = (const int*)  d_in[1];
    const float* tok   = (const float*)d_in[2];
    const float* pos   = (const float*)d_in[3];
    const float* typ   = (const float*)d_in[4];
    const float* elng  = (const float*)d_in[5];
    const float* elnb  = (const float*)d_in[6];
    const float* Wq    = (const float*)d_in[7];
    const float* bq    = (const float*)d_in[8];
    const float* Wk    = (const float*)d_in[9];
    const float* bk    = (const float*)d_in[10];
    const float* Wv    = (const float*)d_in[11];
    const float* bv    = (const float*)d_in[12];
    const float* Wo    = (const float*)d_in[13];
    const float* bo    = (const float*)d_in[14];
    const float* ln1g  = (const float*)d_in[15];
    const float* ln1b  = (const float*)d_in[16];
    const float* W1    = (const float*)d_in[17];
    const float* b1    = (const float*)d_in[18];
    const float* W2    = (const float*)d_in[19];
    const float* b2    = (const float*)d_in[20];
    const float* ln2g  = (const float*)d_in[21];
    const float* ln2b  = (const float*)d_in[22];
    const float* labW  = (const float*)d_in[23];
    const float* labb  = (const float*)d_in[24];
    const float* trans = (const float*)d_in[25];

    // ---- workspace carve-out (256B aligned) ----
    char* ws = (char*)d_ws;
    size_t off = 0;
    auto alloc = [&](size_t bytes) -> char* {
        off = (off + 255) & ~(size_t)255;
        char* p = ws + off;
        off += bytes;
        return p;
    };
    const size_t NTOK = (size_t)B_ * T_;        // 8192
    bf16* wq_t  = (bf16*)alloc((size_t)L_ * H_ * H_ * 2);
    bf16* wk_t  = (bf16*)alloc((size_t)L_ * H_ * H_ * 2);
    bf16* wv_t  = (bf16*)alloc((size_t)L_ * H_ * H_ * 2);
    bf16* wo_t  = (bf16*)alloc((size_t)L_ * H_ * H_ * 2);
    bf16* w1_t  = (bf16*)alloc((size_t)L_ * H_ * FF_ * 2);
    bf16* w2_t  = (bf16*)alloc((size_t)L_ * H_ * FF_ * 2);
    bf16* lw_t  = (bf16*)alloc((size_t)NL_ * H_ * 2);
    float* x    = (float*)alloc(NTOK * H_ * 4);
    bf16*  xb   = (bf16*) alloc(NTOK * H_ * 2);
    float* tmp  = (float*)alloc(NTOK * H_ * 4);
    bf16*  qb   = (bf16*) alloc(NTOK * H_ * 2);
    bf16*  kb   = (bf16*) alloc(NTOK * H_ * 2);
    bf16*  vb   = (bf16*) alloc(NTOK * H_ * 2);
    bf16*  vt   = (bf16*) alloc(NTOK * H_ * 2);
    bf16*  ctxb = (bf16*) alloc(NTOK * H_ * 2);
    // scores (f32, 100MB) and FFN hidden (bf16, 50MB) never live simultaneously:
    char*  big  = alloc((size_t)B_ * NH_ * T_ * T_ * 4);
    float* scores = (float*)big;
    bf16*  hb     = (bf16*) big;
    bf16*  probs  = (bf16*) alloc((size_t)B_ * NH_ * T_ * T_ * 2);
    float* feats  = (float*)alloc(NTOK * NLP_ * 4);
    int*   psi    = (int*)  alloc((size_t)(T_ - 1) * B_ * NL_ * 4);

    // ---- 1. convert + transpose all weights to bf16 [N,K] ----
    {
        dim3 g(((size_t)H_ * H_ + 255) / 256, 1, L_);
        cvt_transpose<<<g, 256, 0, stream>>>(Wq, wq_t, H_, H_, (long long)H_ * H_, (long long)H_ * H_);
        cvt_transpose<<<g, 256, 0, stream>>>(Wk, wk_t, H_, H_, (long long)H_ * H_, (long long)H_ * H_);
        cvt_transpose<<<g, 256, 0, stream>>>(Wv, wv_t, H_, H_, (long long)H_ * H_, (long long)H_ * H_);
        cvt_transpose<<<g, 256, 0, stream>>>(Wo, wo_t, H_, H_, (long long)H_ * H_, (long long)H_ * H_);
    }
    {
        dim3 g(((size_t)H_ * FF_ + 255) / 256, 1, L_);
        cvt_transpose<<<g, 256, 0, stream>>>(W1, w1_t, H_, FF_, (long long)H_ * FF_, (long long)H_ * FF_);
        cvt_transpose<<<g, 256, 0, stream>>>(W2, w2_t, FF_, H_, (long long)H_ * FF_, (long long)H_ * FF_);
    }
    {
        dim3 g(((size_t)H_ * NL_ + 255) / 256, 1, 1);
        cvt_transpose<<<g, 256, 0, stream>>>(labW, lw_t, H_, NL_, 0, 0);
    }

    // ---- 2. embedding + LN ----
    embed_ln<<<(int)NTOK, 256, 0, stream>>>(ids, tok, pos, typ, elng, elnb, x, xb);

    // block tile = 128(M) x 256(N)
    const dim3 gTok768((H_ + 255) / 256, ((int)NTOK + 127) / 128, 1);   // M=8192,N=768
    const dim3 gTokFF ((FF_ + 255) / 256, ((int)NTOK + 127) / 128, 1); // M=8192,N=3072
    const dim3 gScore ((T_ + 255) / 256, (T_ + 127) / 128, B_ * NH_);  // per-head QK^T
    const dim3 gCtx   ((DH_ + 255) / 256, (T_ + 127) / 128, B_ * NH_); // per-head PV

    // ---- 3. encoder layers ----
    for (int l = 0; l < L_; ++l) {
        const bf16* wql = wq_t + (size_t)l * H_ * H_;
        const bf16* wkl = wk_t + (size_t)l * H_ * H_;
        const bf16* wvl = wv_t + (size_t)l * H_ * H_;
        const bf16* wol = wo_t + (size_t)l * H_ * H_;
        const bf16* w1l = w1_t + (size_t)l * H_ * FF_;
        const bf16* w2l = w2_t + (size_t)l * H_ * FF_;

        // QKV projections -> bf16 [B,T,H]
        gemm_bf16<<<gTok768, 256, 0, stream>>>(xb, wql, bq + l * H_, nullptr, qb,
            (int)NTOK, H_, H_, H_, H_, H_, 1, 0,0,0,0,0,0, 1.0f, 0);
        gemm_bf16<<<gTok768, 256, 0, stream>>>(xb, wkl, bk + l * H_, nullptr, kb,
            (int)NTOK, H_, H_, H_, H_, H_, 1, 0,0,0,0,0,0, 1.0f, 0);
        gemm_bf16<<<gTok768, 256, 0, stream>>>(xb, wvl, bv + l * H_, nullptr, vb,
            (int)NTOK, H_, H_, H_, H_, H_, 1, 0,0,0,0,0,0, 1.0f, 0);

        // scores[b,h] = (1/8) * Qh[T,DH] * Kh[T,DH]^T   (z = b*NH + h)
        gemm_bf16<<<gScore, 256, 0, stream>>>(qb, kb, nullptr, scores, nullptr,
            T_, T_, DH_, H_, H_, T_,
            NH_, (long long)T_ * H_, DH_, (long long)T_ * H_, DH_,
            (long long)NH_ * T_ * T_, (long long)T_ * T_, 0.125f, 0);

        softmax_mask<<<B_ * NH_ * T_, 256, 0, stream>>>(scores, msk, probs);

        transpose_v<<<(int)(((size_t)B_ * NH_ * DH_ * T_ + 255) / 256), 256, 0, stream>>>(vb, vt);

        // ctx[b,t,h*DH+d] = P[b,h][T,T] * Vt[b,h][DH,T]^T  -> bf16 [B,T,H]
        gemm_bf16<<<gCtx, 256, 0, stream>>>(probs, vt, nullptr, nullptr, ctxb,
            T_, DH_, T_, T_, T_, H_,
            NH_, (long long)NH_ * T_ * T_, (long long)T_ * T_,
            (long long)NH_ * DH_ * T_, (long long)DH_ * T_,
            (long long)T_ * H_, (long long)DH_, 1.0f, 0);

        // attention output projection
        gemm_bf16<<<gTok768, 256, 0, stream>>>(ctxb, wol, bo + l * H_, tmp, nullptr,
            (int)NTOK, H_, H_, H_, H_, H_, 1, 0,0,0,0,0,0, 1.0f, 0);
        residual_ln<<<(int)NTOK, 256, 0, stream>>>(x, tmp, ln1g + l * H_, ln1b + l * H_, xb);

        // FFN
        gemm_bf16<<<gTokFF, 256, 0, stream>>>(xb, w1l, b1 + l * FF_, nullptr, hb,
            (int)NTOK, FF_, H_, H_, H_, FF_, 1, 0,0,0,0,0,0, 1.0f, /*gelu*/1);
        gemm_bf16<<<gTok768, 256, 0, stream>>>(hb, w2l, b2 + l * H_, tmp, nullptr,
            (int)NTOK, H_, FF_, FF_, FF_, H_, 1, 0,0,0,0,0,0, 1.0f, 0);
        residual_ln<<<(int)NTOK, 256, 0, stream>>>(x, tmp, ln2g + l * H_, ln2b + l * H_, xb);
    }

    // ---- 4. classifier feats (N=31, ldc=32) ----
    {
        dim3 g(1, ((int)NTOK + 127) / 128, 1);
        gemm_bf16<<<g, 256, 0, stream>>>(xb, lw_t, labb, feats, nullptr,
            (int)NTOK, NL_, H_, H_, H_, NLP_, 1, 0,0,0,0,0,0, 1.0f, 0);
    }

    // ---- 5. Viterbi ----
    viterbi_kernel<<<B_, 32, 0, stream>>>(feats, trans, (float*)d_out, psi);
}
